// FusedAGGemm_45664092291685
// MI455X (gfx1250) — compile-verified, matrix operand order checked by reading, not run on hardware
//
#include <hip/hip_runtime.h>

typedef __bf16 v16bf __attribute__((ext_vector_type(16)));
typedef __bf16 bf2   __attribute__((ext_vector_type(2)));
typedef float  v8f   __attribute__((ext_vector_type(8)));

#define M_DIM 4096
#define K_DIM 8192
#define N_DIM 1024
#define KLOC  1024
#define BM 128
#define BN 128
#define BK 32
#define LDSS 40      // padded LDS row stride (elements): 80B, keeps b128 frag reads 16B-aligned
#define NKT (K_DIM / BK)

struct __align__(16) B128 { unsigned int d[4]; };
union FragU { v16bf v; B128 q[2]; };

// Split x into bf16 hi + bf16 lo (x ~= hi + lo, rel err ~2^-16) using HW converts.
__device__ __forceinline__ void split2(float x0, float x1,
                                       unsigned int& hi, unsigned int& lo) {
    bf2 h, l;
    h.x = (__bf16)x0;
    h.y = (__bf16)x1;
    l.x = (__bf16)(x0 - (float)h.x);
    l.y = (__bf16)(x1 - (float)h.y);
    hi = __builtin_bit_cast(unsigned int, h);
    lo = __builtin_bit_cast(unsigned int, l);
}

// 128 threads = 4 waves; each wave owns a 64x64 tile (4x4 WMMA tiles).
__global__ __launch_bounds__(128)
void fused_ag_gemm_bf16x2(const float* __restrict__ A,    // (8, 4096, 1024)
                          const float* __restrict__ W,    // (8192, 1024)
                          const float* __restrict__ bias, // (1024)
                          float* __restrict__ C)          // (4096, 1024)
{
    __shared__ unsigned short Ah[BM][LDSS];
    __shared__ unsigned short Al[BM][LDSS];
    __shared__ unsigned short Bh[BN][LDSS];   // stored n-major (transposed)
    __shared__ unsigned short Bl[BN][LDSS];

    const int tid  = threadIdx.x;
    const int lane = tid & 31;
    const int wid  = tid >> 5;          // 0..3
    const int wm   = (wid & 1) * 64;    // wave M offset in block tile
    const int wn   = (wid >> 1) * 64;   // wave N offset in block tile
    const int bm   = blockIdx.y * BM;
    const int bn   = blockIdx.x * BN;

    // cooperative-load slots (128 threads)
    const int arow = tid >> 3;          // 0..15, +16 per pass (8 passes -> 128 rows)
    const int acol = (tid & 7) * 4;     // 0..28
    const int wk   = wid * 2;           // 0..6 even; thread owns k-rows wk, wk+1 (+8 per pass)
    const int wcol = (tid & 31) * 4;    // 0..124

    float4 aR[8];
    float4 wR[4][2];                    // [pass][row-of-pair]

    auto load_tile = [&](int kt) {
        const int kg = kt * BK;
        const int r  = kg >> 10;            // source rank for this K-tile
        const int kl = kg & (KLOC - 1);     // offset inside that rank's shard
        const float* aB = A + (size_t)r * M_DIM * KLOC + (size_t)bm * KLOC + kl;
        const float* wB = W + (size_t)kg * N_DIM + bn;
#pragma unroll
        for (int p = 0; p < 8; ++p)
            aR[p] = *(const float4*)(aB + (size_t)(arow + 16 * p) * KLOC + acol);
#pragma unroll
        for (int p = 0; p < 4; ++p) {
            const int k = wk + 8 * p;
            wR[p][0] = *(const float4*)(wB + (size_t)k       * N_DIM + wcol);
            wR[p][1] = *(const float4*)(wB + (size_t)(k + 1) * N_DIM + wcol);
        }
    };

    auto stage_tile = [&]() {
        // A: row-major, packed u32 stores (2 bf16 along k)
#pragma unroll
        for (int p = 0; p < 8; ++p) {
            const int row = arow + 16 * p;
            unsigned int h01, l01, h23, l23;
            split2(aR[p].x, aR[p].y, h01, l01);
            split2(aR[p].z, aR[p].w, h23, l23);
            *(unsigned int*)&Ah[row][acol]     = h01;
            *(unsigned int*)&Ah[row][acol + 2] = h23;
            *(unsigned int*)&Al[row][acol]     = l01;
            *(unsigned int*)&Al[row][acol + 2] = l23;
        }
        // W: transpose to n-major; pack two consecutive k into one u32 store
#pragma unroll
        for (int p = 0; p < 4; ++p) {
            const int k = wk + 8 * p;
            float r0[4] = {wR[p][0].x, wR[p][0].y, wR[p][0].z, wR[p][0].w};
            float r1[4] = {wR[p][1].x, wR[p][1].y, wR[p][1].z, wR[p][1].w};
#pragma unroll
            for (int j = 0; j < 4; ++j) {
                unsigned int h, l;
                split2(r0[j], r1[j], h, l);          // (k, k+1) for column n=wcol+j
                *(unsigned int*)&Bh[wcol + j][k] = h;
                *(unsigned int*)&Bl[wcol + j][k] = l;
            }
        }
    };

    v8f acc[4][4];
#pragma unroll
    for (int i = 0; i < 4; ++i)
#pragma unroll
        for (int j = 0; j < 4; ++j)
            acc[i][j] = (v8f){0.f, 0.f, 0.f, 0.f, 0.f, 0.f, 0.f, 0.f};

    // ISA fragment lane mapping: row/col = lane&15, lane>>4 selects K-half
    const int fr    = lane & 15;
    const int khalf = lane >> 4;
    const int coff  = khalf * 8;

    load_tile(0);
    for (int kt = 0; kt < NKT; ++kt) {
        stage_tile();
        __syncthreads();
        if (kt + 1 < NKT) load_tile(kt + 1);   // overlap next global fetch with WMMA

        // Load all 8 B fragments once (reused by 4 m-tiles -> 1.5 WMMA per ds_load_b128)
        FragU fbh[4], fbl[4];
#pragma unroll
        for (int j = 0; j < 4; ++j) {
            const unsigned short* pb = &Bh[wn + j * 16 + fr][coff];
            fbh[j].q[0] = *(const B128*)pb;
            fbh[j].q[1] = *(const B128*)(pb + 16);
            const unsigned short* pc = &Bl[wn + j * 16 + fr][coff];
            fbl[j].q[0] = *(const B128*)pc;
            fbl[j].q[1] = *(const B128*)(pc + 16);
        }
#pragma unroll
        for (int i = 0; i < 4; ++i) {
            FragU fah, fal;
            const unsigned short* pa = &Ah[wm + i * 16 + fr][coff];
            fah.q[0] = *(const B128*)pa;
            fah.q[1] = *(const B128*)(pa + 16);
            const unsigned short* pl = &Al[wm + i * 16 + fr][coff];
            fal.q[0] = *(const B128*)pl;
            fal.q[1] = *(const B128*)(pl + 16);
#pragma unroll
            for (int j = 0; j < 4; ++j) {
                // A*W ~= Ah*Wh + Ah*Wl + Al*Wh  (fp32 accumulate, ~1e-5 rel err)
                acc[i][j] = __builtin_amdgcn_wmma_f32_16x16x32_bf16(
                    false, fah.v, false, fbh[j].v, (short)0, acc[i][j], false, false);
                acc[i][j] = __builtin_amdgcn_wmma_f32_16x16x32_bf16(
                    false, fah.v, false, fbl[j].v, (short)0, acc[i][j], false, false);
                acc[i][j] = __builtin_amdgcn_wmma_f32_16x16x32_bf16(
                    false, fal.v, false, fbh[j].v, (short)0, acc[i][j], false, false);
            }
        }
        __syncthreads();
    }

    // Epilogue: C/D layout — VGPR e: lanes 0-15 hold M=e, lanes 16-31 hold M=8+e; N = lane&15
#pragma unroll
    for (int j = 0; j < 4; ++j) {
        const int cn = bn + wn + j * 16 + fr;
        const float bv = bias[cn];
#pragma unroll
        for (int i = 0; i < 4; ++i) {
            const int rbase = bm + wm + i * 16 + khalf * 8;
#pragma unroll
            for (int e = 0; e < 8; ++e)
                C[(size_t)(rbase + e) * N_DIM + cn] = acc[i][j][e] + bv;
        }
    }
}

extern "C" void kernel_launch(void* const* d_in, const int* in_sizes, int n_in,
                              void* d_out, int out_size, void* d_ws, size_t ws_size,
                              hipStream_t stream) {
    const float* A    = (const float*)d_in[0];   // all_act (8, 4096, 1024)
    const float* W    = (const float*)d_in[1];   // local_W (8192, 1024)
    const float* bias = (const float*)d_in[2];   // bias (1024)
    float* C          = (float*)d_out;           // (4096, 1024)

    dim3 grid(N_DIM / BN, M_DIM / BM);           // (8, 32) = 256 workgroups, 2 per WGP
    fused_ag_gemm_bf16x2<<<grid, 128, 0, stream>>>(A, W, bias, C);
}
